// CapsuleNet_60455959658893
// MI455X (gfx1250) — compile-verified, matrix operand order checked by reading, not run on hardware
//
#include <hip/hip_runtime.h>
#include <hip/hip_bf16.h>
#include <math.h>

typedef _Float16 v8h  __attribute__((ext_vector_type(8)));
typedef _Float16 v16h __attribute__((ext_vector_type(16)));
typedef float    v2f  __attribute__((ext_vector_type(2)));
typedef float    v8f  __attribute__((ext_vector_type(8)));

#if defined(__has_builtin)
#if __has_builtin(__builtin_amdgcn_tensor_load_to_lds) && __has_builtin(__builtin_amdgcn_s_wait_tensorcnt)
#define HAS_TDM 1
#endif
#endif
#ifndef HAS_TDM
#define HAS_TDM 0
#endif

#if HAS_TDM
typedef unsigned int u32x4 __attribute__((ext_vector_type(4)));
typedef int          i32x4 __attribute__((ext_vector_type(4)));
typedef int          i32x8 __attribute__((ext_vector_type(8)));
#endif

#define LDA 40   // 32 halves + 8 pad -> 80B row stride, conflict-free b128 LDS reads

__device__ __forceinline__ v16h load_frag16(const _Float16* p) {
    v8h lo = *(const v8h*)(p);
    v8h hi = *(const v8h*)(p + 16);
    return __builtin_shufflevector(lo, hi, 0,1,2,3,4,5,6,7,8,9,10,11,12,13,14,15);
}

// ---------------------------------------------------------------------------
// Weight prep
// ---------------------------------------------------------------------------
__global__ void prep_w1(const float* __restrict__ w, _Float16* __restrict__ w1h) {
    int t = blockIdx.x * 256 + threadIdx.x;           // 256*96
    if (t >= 256 * 96) return;
    int n = t / 96, k = t % 96;
    w1h[t] = (k < 81) ? (_Float16)w[n * 81 + k] : (_Float16)0.f;
}

// conv2_w (256,256,9,9) f32 -> w2h[n][tap*256+ic] f16  (tap = kh*9+kw)
__global__ void prep_w2(const float* __restrict__ w, _Float16* __restrict__ w2h) {
    int t = blockIdx.x * 256 + threadIdx.x;           // 256*20736
    if (t >= 256 * 20736) return;
    int n = t / 20736, rem = t % 20736;
    int tap = rem / 256, ic = rem % 256;
    w2h[t] = (_Float16)w[(size_t)n * 20736 + ic * 81 + tap];
}

// ---------------------------------------------------------------------------
// conv1 implicit GEMM:  M=230400 (b,y,x), N=256, K=96.  out: h NHWC f16 (ReLU)
// BM=128 BN=128 BK=32, 8 waves (2x4), wave tile 64x32
// ---------------------------------------------------------------------------
__global__ __launch_bounds__(256) void conv1_gemm(const float* __restrict__ x,
        const _Float16* __restrict__ w1h, const float* __restrict__ bias,
        _Float16* __restrict__ h) {
    __shared__ _Float16 As[128 * LDA];
    __shared__ _Float16 Bs[128 * LDA];
    __shared__ int rowBase[128];

    int bx = blockIdx.x;
    int mt = bx >> 1, nt = bx & 1;
    int m0 = mt * 128, n0 = nt * 128;
    int tid = threadIdx.x;
    if (tid < 128) {
        int m = m0 + tid;
        int b = m / 14400, rem = m % 14400;
        int y = rem / 120, xx = rem % 120;
        rowBase[tid] = (b * 128 + y) * 128 + xx;
    }
    int lane = tid & 31, w = tid >> 5;
    int mw = (w >> 2) * 64;
    int nw = (w & 3) * 32;

    v8f acc[4][2] = {};
    for (int kt = 0; kt < 3; ++kt) {
        __syncthreads();
        for (int i = tid; i < 128 * 32; i += 256) {
            int row = i >> 5, kk = i & 31;
            int k = kt * 32 + kk;
            _Float16 val = (_Float16)0.f;
            if (k < 81) {
                int kh = k / 9, kw = k - kh * 9;
                val = (_Float16)x[rowBase[row] + kh * 128 + kw];
            }
            As[row * LDA + kk] = val;
        }
        for (int i = tid; i < 512; i += 256) {
            int row = i >> 2, seg = i & 3;
            *(uint4*)&Bs[row * LDA + seg * 8] =
                *(const uint4*)&w1h[(n0 + row) * 96 + kt * 32 + seg * 8];
        }
        __syncthreads();

        int kbase = (lane < 16) ? 0 : 8;
        v16h af[4], bf[2];
#pragma unroll
        for (int mf = 0; mf < 4; ++mf)
            af[mf] = load_frag16(&As[(mw + mf * 16 + (lane & 15)) * LDA + kbase]);
#pragma unroll
        for (int nf = 0; nf < 2; ++nf)
            bf[nf] = load_frag16(&Bs[(nw + nf * 16 + (lane & 15)) * LDA + kbase]);
#pragma unroll
        for (int mf = 0; mf < 4; ++mf)
#pragma unroll
            for (int nf = 0; nf < 2; ++nf)
                acc[mf][nf] = __builtin_amdgcn_wmma_f32_16x16x32_f16(
                    false, af[mf], false, bf[nf], (short)0, acc[mf][nf], false, false);
    }
#pragma unroll
    for (int mf = 0; mf < 4; ++mf)
#pragma unroll
        for (int nf = 0; nf < 2; ++nf) {
            int n = n0 + nw + nf * 16 + (lane & 15);
            float bn = bias[n];
#pragma unroll
            for (int v = 0; v < 8; ++v) {
                int m = m0 + mw + mf * 16 + v + ((lane >> 4) << 3);
                float val = acc[mf][nf][v] + bn;
                h[(size_t)m * 256 + n] = (_Float16)fmaxf(val, 0.f);
            }
        }
}

// ---------------------------------------------------------------------------
// conv2 implicit GEMM:  M=50176, N=256, K=20736 (k = tap*256+ic). out p f32 NHWC.
// BM=112 (= 2 output rows x 56): A-tile is a regular TDM 3D tile
//   tile dims (32 ic, 56 x, 2 y), elem strides (1, 512, 61440), f16.
// B-tile: 2D (32 k, 256 oc rows), stride 20736.
// Double-buffered LDS; TDM (tensor_load_to_lds) issued by wave 0, synced via
// s_wait_tensorcnt + workgroup barrier. pad_interval=16dw/pad_amount=4dw
// reproduces the LDA=40 row layout. 8 waves (1x8), wave tile 112x32.
// ---------------------------------------------------------------------------
#if HAS_TDM
__device__ __forceinline__ void tdm_issue(unsigned int lds, unsigned long long ga,
                                          i32x8 g1, i32x4 g2, i32x4 g3) {
    u32x4 g0;
    g0[0] = 1u;                                   // count=1, user mode
    g0[1] = lds;                                  // lds_addr (bytes)
    g0[2] = (unsigned int)ga;                     // global_addr[31:0]
    g0[3] = (unsigned int)((ga >> 32) & 0x1FFFFFFu) | (2u << 30);  // [56:32] | type=2
    i32x8 gx = { 0, 0, 0, 0, 0, 0, 0, 0 };        // unused trailing group (VADDR4 slot)
    __builtin_amdgcn_tensor_load_to_lds(g0, g1, g2, g3, gx, 0);
}
#endif

__global__ __launch_bounds__(256) void conv2_gemm(const _Float16* __restrict__ h,
        const _Float16* __restrict__ w2h, const float* __restrict__ bias,
        float* __restrict__ p) {
    int tid = threadIdx.x;
    int lane = tid & 31, w = tid >> 5;
    int m0 = blockIdx.x * 112;
    int bb = blockIdx.x / 28;          // batch
    int tt = blockIdx.x % 28;          // y-pair tile: output rows 2*tt, 2*tt+1
    int nw = w * 32;

    v8f acc[7][2] = {};

#if HAS_TDM
    __shared__ _Float16 As[2][112 * LDA];
    __shared__ _Float16 Bs[2][256 * LDA];

    // ---- invariant descriptor groups (ISA 8.4-8.6 bit layout) ----
    // flags: data_size=2B(1), pad_enable, pad_interval=16dw(3), pad_amount=4dw(3)
    const unsigned int FLAGS = (1u << 16) | (1u << 20) | (3u << 22) | (3u << 25);
    i32x8 g1A, g1B;
    g1A[0] = (int)FLAGS;
    g1A[1] = (int)0xFFFF0000u;                 // tensor_dim0 lo16 (huge dims: no OOB)
    g1A[2] = (int)0xFFFF0FFFu;                 // tensor_dim0 hi / tensor_dim1 lo
    g1A[3] = (int)(0x0FFFu | (32u << 16));     // tensor_dim1 hi | tile_dim0=32
    g1A[4] = (int)(56u | (2u << 16));          // tile_dim1=56 | tile_dim2=2
    g1A[5] = 512;                              // tensor_dim0_stride (x step, elems)
    g1A[6] = (int)0xF0000000u;                 // stride0 hi16=0 | stride1 lo16 (61440)
    g1A[7] = 0;                                // stride1 hi32 (61440>>16 == 0)
    g1B[0] = (int)FLAGS;
    g1B[1] = (int)0xFFFF0000u;
    g1B[2] = (int)0xFFFF0FFFu;
    g1B[3] = (int)(0x0FFFu | (32u << 16));
    g1B[4] = 256;                              // tile_dim1=256 rows, tile_dim2=0
    g1B[5] = 20736;                            // row stride (elems)
    g1B[6] = 0;
    g1B[7] = 0;
    i32x4 g2; g2[0] = (int)0x0FFFFFFFu; g2[1] = 0; g2[2] = 0; g2[3] = 0;  // tensor_dim2 huge
    i32x4 g3; g3[0] = 0; g3[1] = 0; g3[2] = 0; g3[3] = 0;

    unsigned int ldsA[2] = { (unsigned int)(uintptr_t)&As[0][0],
                             (unsigned int)(uintptr_t)&As[1][0] };
    unsigned int ldsB[2] = { (unsigned int)(uintptr_t)&Bs[0][0],
                             (unsigned int)(uintptr_t)&Bs[1][0] };
    unsigned long long baseA = (unsigned long long)(uintptr_t)h;
    unsigned long long baseB = (unsigned long long)(uintptr_t)w2h;

    auto issue = [&](int kt, int buf) {
        int tap = kt >> 3;                       // 8 k-chunks per tap
        int icb = (kt & 7) * 32;
        int kh = tap / 9, kw2 = tap - kh * 9;
        unsigned long long gaA = baseA +
            2ull * ((unsigned long long)(((bb * 120 + 4 * tt + kh) * 120 + kw2)) * 256 + icb);
        unsigned long long gaB = baseB + 2ull * ((unsigned long long)kt * 32);
        tdm_issue(ldsA[buf], gaA, g1A, g2, g3);
        tdm_issue(ldsB[buf], gaB, g1B, g2, g3);
    };

    if (w == 0) issue(0, 0);
    for (int kt = 0; kt < 648; ++kt) {
        int cur = kt & 1;
        if (w == 0) {
            if (kt + 1 < 648) {
                issue(kt + 1, cur ^ 1);
                __builtin_amdgcn_s_wait_tensorcnt((short)2);   // pair kt complete
            } else {
                __builtin_amdgcn_s_wait_tensorcnt((short)0);
            }
        }
        __syncthreads();                        // buffer `cur` visible to all waves
        const _Float16* Ab = &As[cur][0];
        const _Float16* Bb = &Bs[cur][0];
        int kbase = (lane < 16) ? 0 : 8;
        v16h af[7], bf[2];
#pragma unroll
        for (int mf = 0; mf < 7; ++mf)
            af[mf] = load_frag16(&Ab[(mf * 16 + (lane & 15)) * LDA + kbase]);
#pragma unroll
        for (int nf = 0; nf < 2; ++nf)
            bf[nf] = load_frag16(&Bb[(nw + nf * 16 + (lane & 15)) * LDA + kbase]);
#pragma unroll
        for (int mf = 0; mf < 7; ++mf)
#pragma unroll
            for (int nf = 0; nf < 2; ++nf)
                acc[mf][nf] = __builtin_amdgcn_wmma_f32_16x16x32_f16(
                    false, af[mf], false, bf[nf], (short)0, acc[mf][nf], false, false);
        __syncthreads();                        // all reads of `cur` done before reuse
    }
#else  // ------- fallback: manual single-buffer staging -------
    __shared__ _Float16 As1[112 * LDA];
    __shared__ _Float16 Bs1[256 * LDA];
    __shared__ int rowBase[112];
    if (tid < 112) {
        int m = m0 + tid;
        int rem = m % 3136;
        int y = rem / 56, xx = rem % 56;
        rowBase[tid] = ((bb * 120 + 2 * y) * 120 + 2 * xx) * 256;
    }
    for (int kt = 0; kt < 648; ++kt) {
        int tap = kt >> 3;
        int icb = (kt & 7) * 32;
        int kh = tap / 9, kw2 = tap - kh * 9;
        int off = (kh * 120 + kw2) * 256 + icb;
        __syncthreads();
        for (int i = tid; i < 448; i += 256) {
            int row = i >> 2, seg = i & 3;
            *(uint4*)&As1[row * LDA + seg * 8] =
                *(const uint4*)&h[(size_t)rowBase[row] + off + seg * 8];
        }
        for (int i = tid; i < 1024; i += 256) {
            int row = i >> 2, seg = i & 3;
            *(uint4*)&Bs1[row * LDA + seg * 8] =
                *(const uint4*)&w2h[(size_t)row * 20736 + kt * 32 + seg * 8];
        }
        __syncthreads();
        int kbase = (lane < 16) ? 0 : 8;
        v16h af[7], bf[2];
#pragma unroll
        for (int mf = 0; mf < 7; ++mf)
            af[mf] = load_frag16(&As1[(mf * 16 + (lane & 15)) * LDA + kbase]);
#pragma unroll
        for (int nf = 0; nf < 2; ++nf)
            bf[nf] = load_frag16(&Bs1[(nw + nf * 16 + (lane & 15)) * LDA + kbase]);
#pragma unroll
        for (int mf = 0; mf < 7; ++mf)
#pragma unroll
            for (int nf = 0; nf < 2; ++nf)
                acc[mf][nf] = __builtin_amdgcn_wmma_f32_16x16x32_f16(
                    false, af[mf], false, bf[nf], (short)0, acc[mf][nf], false, false);
    }
#endif

#pragma unroll
    for (int mf = 0; mf < 7; ++mf)
#pragma unroll
        for (int nf = 0; nf < 2; ++nf) {
            int n = nw + nf * 16 + (lane & 15);
            float bn = bias[n];
#pragma unroll
            for (int v = 0; v < 8; ++v) {
                int m = m0 + mf * 16 + v + ((lane >> 4) << 3);
                p[(size_t)m * 256 + n] = acc[mf][nf][v] + bn;
            }
        }
}

// ---------------------------------------------------------------------------
// squash primary capsules: p NHWC f32 -> u[r][b][8] f32 (r = s*32+g)
// ---------------------------------------------------------------------------
__global__ void squash_u(const float* __restrict__ p, float* __restrict__ u) {
    int t = blockIdx.x * 256 + threadIdx.x;      // t = b*R + r
    if (t >= 16 * 100352) return;
    int b = t / 100352, r = t % 100352;
    int s = r >> 5, g = r & 31;
    const float* src = p + ((size_t)b * 3136 + s) * 256 + g * 8;
    float e[8], n2 = 0.f;
#pragma unroll
    for (int i = 0; i < 8; ++i) { e[i] = src[i]; n2 += e[i] * e[i]; }
    float nrm = sqrtf(n2);
    float sc = (n2 / (1.f + n2)) / (nrm + 1e-8f);
    float* dst = u + ((size_t)r * 16 + b) * 8;
#pragma unroll
    for (int i = 0; i < 8; ++i) dst[i] = e[i] * sc;
}

// ---------------------------------------------------------------------------
// routing pass: u_hat per capsule via f32 WMMA 16x16x4 (M=b, N=(c,o), K=8)
// ---------------------------------------------------------------------------
__global__ __launch_bounds__(256) void route_pass(const float* __restrict__ u,
        const float* __restrict__ rw, const float* __restrict__ va,
        const float* __restrict__ vb, float* __restrict__ partials, int iter) {
    __shared__ float shV[512];
    int tid = threadIdx.x;
    if (iter > 0) {
        for (int i = tid; i < 512; i += 256)
            shV[i] = va[i] + ((iter == 2) ? vb[i] : 0.f);
    }
    __syncthreads();

    int lane = tid & 31, w = tid >> 5;
    int gid = blockIdx.x * 8 + w;
    int r0 = gid * 49;
    int bb0 = lane & 15;
    int o = lane & 15;
    int kb = (lane < 16) ? 0 : 2;
    int hb = (lane >> 4) << 3;

    v8f accS0 = {}, accS1 = {};
    for (int rr = 0; rr < 49; ++rr) {
        int r = r0 + rr;
        const float* up = u + ((size_t)r * 16 + bb0) * 8;
        v2f aLo = { up[kb],     up[kb + 1] };
        v2f aHi = { up[kb + 4], up[kb + 5] };

        const float* w0p = rw + (size_t)r * 256;
        const float* w1p = w0p + 128;
        v2f bLo0 = { w0p[kb * 16 + o],       w0p[(kb + 1) * 16 + o] };
        v2f bHi0 = { w0p[(kb + 4) * 16 + o], w0p[(kb + 5) * 16 + o] };
        v2f bLo1 = { w1p[kb * 16 + o],       w1p[(kb + 1) * 16 + o] };
        v2f bHi1 = { w1p[(kb + 4) * 16 + o], w1p[(kb + 5) * 16 + o] };

        v8f z = {};
        v8f uh0 = __builtin_amdgcn_wmma_f32_16x16x4_f32(false, aLo, false, bLo0, (short)0, z,   false, false);
        uh0     = __builtin_amdgcn_wmma_f32_16x16x4_f32(false, aHi, false, bHi0, (short)0, uh0, false, false);
        v8f uh1 = __builtin_amdgcn_wmma_f32_16x16x4_f32(false, aLo, false, bLo1, (short)0, z,   false, false);
        uh1     = __builtin_amdgcn_wmma_f32_16x16x4_f32(false, aHi, false, bHi1, (short)0, uh1, false, false);

        if (iter == 0) {
            accS0 += uh0 * 0.5f;
            accS1 += uh1 * 0.5f;
        } else {
#pragma unroll
            for (int v = 0; v < 8; ++v) {
                int bbat = v + hb;
                float t0 = uh0[v] * shV[(bbat * 2 + 0) * 16 + o];
                float t1 = uh1[v] * shV[(bbat * 2 + 1) * 16 + o];
#pragma unroll
                for (int m = 1; m < 16; m <<= 1) {
                    t0 += __shfl_xor(t0, m, 32);
                    t1 += __shfl_xor(t1, m, 32);
                }
                float mx = fmaxf(t0, t1);
                float e0 = expf(t0 - mx), e1 = expf(t1 - mx);
                float inv = 1.f / (e0 + e1);
                accS0[v] += e0 * inv * uh0[v];
                accS1[v] += e1 * inv * uh1[v];
            }
        }
    }
    size_t base = (size_t)gid * 512;
#pragma unroll
    for (int v = 0; v < 8; ++v) {
        partials[base + v * 32 + lane]       = accS0[v];
        partials[base + 256 + v * 32 + lane] = accS1[v];
    }
}

__global__ __launch_bounds__(512) void route_reduce(const float* __restrict__ partials,
        float* __restrict__ vout) {
    __shared__ float sh[512];
    int t = threadIdx.x;
    float s = 0.f;
    for (int g = 0; g < 2048; ++g) s += partials[(size_t)g * 512 + t];
    int c = t >> 8, v = (t >> 5) & 7, lane = t & 31;
    int b = v + ((lane >> 4) << 3), o = lane & 15;
    sh[(c * 16 + b) * 16 + o] = s;
    __syncthreads();
    float n2 = 0.f;
    for (int oo = 0; oo < 16; ++oo) {
        float x = sh[(c * 16 + b) * 16 + oo];
        n2 += x * x;
    }
    float nrm = sqrtf(n2);
    vout[(b * 2 + c) * 16 + o] = (n2 / (1.f + n2)) * (s / (nrm + 1e-8f));
}

// ---------------------------------------------------------------------------
// head MLP: flat(16x32) -> 128 -> 64 -> 2, log_softmax
// ---------------------------------------------------------------------------
__global__ __launch_bounds__(128) void head(const float* __restrict__ v3,
        const float* __restrict__ w1, const float* __restrict__ b1,
        const float* __restrict__ w2, const float* __restrict__ b2,
        const float* __restrict__ w3, const float* __restrict__ b3,
        float* __restrict__ out) {
    __shared__ float fl[16 * 32];
    __shared__ float h1[16 * 128];
    __shared__ float h2[16 * 64];
    int t = threadIdx.x;
    for (int i = t; i < 512; i += 128) fl[i] = v3[i];
    __syncthreads();
    for (int b = 0; b < 16; ++b) {
        float s = b1[t];
        for (int j = 0; j < 32; ++j) s += fl[b * 32 + j] * w1[j * 128 + t];
        h1[b * 128 + t] = fmaxf(s, 0.f);
    }
    __syncthreads();
    if (t < 64)
        for (int b = 0; b < 16; ++b) {
            float s = b2[t];
            for (int k = 0; k < 128; ++k) s += h1[b * 128 + k] * w2[k * 64 + t];
            h2[b * 64 + t] = fmaxf(s, 0.f);
        }
    __syncthreads();
    if (t < 16) {
        int b = t;
        float l0 = b3[0], l1 = b3[1];
        for (int k = 0; k < 64; ++k) {
            l0 += h2[b * 64 + k] * w3[k * 2 + 0];
            l1 += h2[b * 64 + k] * w3[k * 2 + 1];
        }
        float mx = fmaxf(l0, l1);
        float lse = mx + logf(expf(l0 - mx) + expf(l1 - mx));
        out[b * 2 + 0] = l0 - lse;
        out[b * 2 + 1] = l1 - lse;
    }
}

// ---------------------------------------------------------------------------
extern "C" void kernel_launch(void* const* d_in, const int* in_sizes, int n_in,
                              void* d_out, int out_size, void* d_ws, size_t ws_size,
                              hipStream_t stream) {
    const float* x       = (const float*)d_in[0];
    const float* conv1_w = (const float*)d_in[1];
    const float* conv1_b = (const float*)d_in[2];
    const float* conv2_w = (const float*)d_in[3];
    const float* conv2_b = (const float*)d_in[4];
    const float* route_w = (const float*)d_in[5];
    const float* w1 = (const float*)d_in[6];
    const float* b1 = (const float*)d_in[7];
    const float* w2 = (const float*)d_in[8];
    const float* b2 = (const float*)d_in[9];
    const float* w3 = (const float*)d_in[10];
    const float* b3 = (const float*)d_in[11];
    float* out = (float*)d_out;

    char* ws = (char*)d_ws;
    _Float16* w1h      = (_Float16*)(ws + 0);                    //   49,152 B
    _Float16* w2h      = (_Float16*)(ws + 49152);                // 10,616,832 B
    _Float16* h        = (_Float16*)(ws + 10665984);             // 117,964,800 B
    float*    p        = (float*)   (ws + 128630784);            // 51,380,224 B
    float*    u        = (float*)   (ws + 180011008);            // 51,380,224 B
    float*    partials = (float*)   (ws + 231391232);            // 4,194,304 B
    float*    v0       = (float*)   (ws + 235585536);            // 2048 B
    float*    v1buf    = (float*)   (ws + 235587584);            // 2048 B
    float*    v2buf    = (float*)   (ws + 235589632);            // 2048 B

    prep_w1<<<96, 256, 0, stream>>>(conv1_w, w1h);
    prep_w2<<<20736, 256, 0, stream>>>(conv2_w, w2h);
    conv1_gemm<<<3600, 256, 0, stream>>>(x, w1h, conv1_b, h);
    conv2_gemm<<<448, 256, 0, stream>>>(h, w2h, conv2_b, p);
    squash_u<<<6272, 256, 0, stream>>>(p, u);

    route_pass<<<256, 256, 0, stream>>>(u, route_w, v0, v0, partials, 0);
    route_reduce<<<1, 512, 0, stream>>>(partials, v0);
    route_pass<<<256, 256, 0, stream>>>(u, route_w, v0, v0, partials, 1);
    route_reduce<<<1, 512, 0, stream>>>(partials, v1buf);
    route_pass<<<256, 256, 0, stream>>>(u, route_w, v0, v1buf, partials, 2);
    route_reduce<<<1, 512, 0, stream>>>(partials, v2buf);

    head<<<1, 128, 0, stream>>>(v2buf, w1, b1, w2, b2, w3, b3, out);
}
